// GQA_60146722013363
// MI455X (gfx1250) — compile-verified, matrix operand order checked by reading, not run on hardware
//
#include <hip/hip_runtime.h>

// ---------------------------------------------------------------------------
// GQA attention block for MI455X (gfx1250, wave32, WMMA 16x16x32 bf16).
//   1) gemm_wmma<float,bf16> : q,k,v projections (bf16 out), 128x128 tiles,
//      register double-buffered LDS staging.
//   2) rope_kernel           : in-place RoPE; folds softmax scale*log2(e)
//                              into Q so attention can use exp2 directly.
//   3) attn_kernel           : causal flash-attention, 64-key tiles,
//                              diagonal-only masking, wave-uniform tile skip,
//                              DPP row reductions, async K-tile loads to LDS.
//   4) gemm_wmma<bf16,float> : out = attn@wo (f32 out).
// ---------------------------------------------------------------------------

typedef __bf16 bf16_t;
typedef __attribute__((ext_vector_type(4)))  bf16_t bf16x4;
typedef __attribute__((ext_vector_type(8)))  bf16_t bf16x8;
typedef __attribute__((ext_vector_type(16))) bf16_t v16bf;
typedef __attribute__((ext_vector_type(8)))  float  v8f;
typedef __attribute__((ext_vector_type(4)))  int    v4i_t;

#define B_  2
#define S_  2048
#define D_  2048
#define HQ_ 32
#define HKV_ 8
#define HD_ 64

#if defined(__has_builtin)
#if __has_builtin(__builtin_amdgcn_global_load_async_to_lds_b128) && \
    __has_builtin(__builtin_amdgcn_s_wait_asynccnt)
#define USE_ASYNC_LDS 1
#endif
#endif

#if defined(USE_ASYNC_LDS)
// Async global->LDS copy (GLOBAL_LOAD_ASYNC_TO_LDS_B128, tracked by ASYNCcnt).
// This toolchain's builtin takes generic pointers to int-vec4 (per the
// compiler diagnostic), plus immediate offset and cache-policy operands.
__device__ __forceinline__ void async_ld_b128(const void* g, void* l) {
  __builtin_amdgcn_global_load_async_to_lds_b128((v4i_t*)g, (v4i_t*)l, 0, 0);
}
#endif

__device__ __forceinline__ v16bf cat16(bf16x8 lo, bf16x8 hi) {
  return __builtin_shufflevector(lo, hi, 0, 1, 2, 3, 4, 5, 6, 7,
                                 8, 9, 10, 11, 12, 13, 14, 15);
}

// A-style WMMA bf16 fragment (row = lane%16; lanes 0-15 take K {0..7,16..23},
// lanes 16-31 take K {8..15,24..31}) from a K-contiguous run of 32 bf16.
__device__ __forceinline__ v16bf load_frag(const bf16_t* rowp, int half) {
  bf16x8 lo = *(const bf16x8*)(rowp + half * 8);
  bf16x8 hi = *(const bf16x8*)(rowp + 16 + half * 8);
  return cat16(lo, hi);
}

// ---- 16-lane row reductions via DPP (VALU, keeps the DS pipe free) --------
template <int CTRL>
__device__ __forceinline__ float dpp_mov(float x) {
  return __builtin_bit_cast(
      float, __builtin_amdgcn_update_dpp(0, __builtin_bit_cast(int, x), CTRL,
                                         0xF, 0xF, true));
}
__device__ __forceinline__ float row_max16(float r) {
  r = fmaxf(r, dpp_mov<0xB1>(r));   // quad_perm [1,0,3,2]  (xor 1)
  r = fmaxf(r, dpp_mov<0x4E>(r));   // quad_perm [2,3,0,1]  (xor 2)
  r = fmaxf(r, dpp_mov<0x124>(r));  // row_ror:4
  r = fmaxf(r, dpp_mov<0x128>(r));  // row_ror:8
  return r;
}
__device__ __forceinline__ float row_sum16(float r) {
  r += dpp_mov<0xB1>(r);
  r += dpp_mov<0x4E>(r);
  r += dpp_mov<0x124>(r);
  r += dpp_mov<0x128>(r);
  return r;
}

// ---------------------------------------------------------------------------
// WMMA GEMM: C[M,N] = A[M,K] * W[K,N].  Block 256 thr = 8 waves.
// Tile 128x128, K-step 32. Wave grid 4x2; each wave: 2x4 C tiles (16x16).
// Global loads for tile k+1 are held in registers while tile k computes.
// ---------------------------------------------------------------------------
template <typename TA, typename TC>
__global__ __launch_bounds__(256) void gemm_wmma(const TA* __restrict__ A,
                                                 const float* __restrict__ W,
                                                 TC* __restrict__ C,
                                                 int M, int N, int K) {
  constexpr int BM = 128, BN = 128, BK = 32;
  __shared__ __align__(16) bf16_t Asm[BM * BK];
  __shared__ __align__(16) bf16_t Bsm[BN * BK];  // transposed: [n][k]

  const int tid  = threadIdx.x;
  const int lane = tid & 31;
  const int wave = tid >> 5;  // 0..7
  const int wm   = wave >> 1; // 0..3 -> 32 rows each
  const int wn   = wave & 1;  // 0..1 -> 64 cols each
  const int l16  = lane & 15;
  const int half = lane >> 4;

  const int m0 = blockIdx.y * BM;
  const int n0 = blockIdx.x * BN;

  v8f acc[2][4] = {};

  float4 arf[4];
  bf16x4 arh[4];
  float4 brf[4];

  auto load_regs = [&](int k0) {
    #pragma unroll
    for (int i = 0; i < 4; ++i) {  // A tile: 128x32 = 4096 elems
      int e = i * 1024 + tid * 4;
      int r = e >> 5, c = e & 31;
      if constexpr (sizeof(TA) == 4)
        arf[i] = *(const float4*)&A[(size_t)(m0 + r) * K + k0 + c];
      else
        arh[i] = *(const bf16x4*)&A[(size_t)(m0 + r) * K + k0 + c];
    }
    #pragma unroll
    for (int i = 0; i < 4; ++i) {  // B tile: 32x128 = 4096 elems
      int e = i * 1024 + tid * 4;
      int kk = e >> 7, nn = e & 127;
      brf[i] = *(const float4*)&W[(size_t)(k0 + kk) * N + n0 + nn];
    }
  };
  auto store_lds = [&]() {
    #pragma unroll
    for (int i = 0; i < 4; ++i) {
      int e = i * 1024 + tid * 4;
      int r = e >> 5, c = e & 31;
      if constexpr (sizeof(TA) == 4) {
        bf16x4 bv = {(bf16_t)arf[i].x, (bf16_t)arf[i].y, (bf16_t)arf[i].z,
                     (bf16_t)arf[i].w};
        *(bf16x4*)&Asm[r * BK + c] = bv;
      } else {
        *(bf16x4*)&Asm[r * BK + c] = arh[i];
      }
    }
    #pragma unroll
    for (int i = 0; i < 4; ++i) {
      int e = i * 1024 + tid * 4;
      int kk = e >> 7, nn = e & 127;
      Bsm[(nn + 0) * BK + kk] = (bf16_t)brf[i].x;
      Bsm[(nn + 1) * BK + kk] = (bf16_t)brf[i].y;
      Bsm[(nn + 2) * BK + kk] = (bf16_t)brf[i].z;
      Bsm[(nn + 3) * BK + kk] = (bf16_t)brf[i].w;
    }
  };

  load_regs(0);
  for (int k0 = 0; k0 < K; k0 += BK) {
    store_lds();
    __syncthreads();
    if (k0 + BK < K) load_regs(k0 + BK);  // overlap with WMMA below

    v16bf af[2];
    #pragma unroll
    for (int mt = 0; mt < 2; ++mt)
      af[mt] = load_frag(&Asm[(wm * 32 + mt * 16 + l16) * BK], half);
    #pragma unroll
    for (int nt = 0; nt < 4; ++nt) {
      v16bf bv = load_frag(&Bsm[(wn * 64 + nt * 16 + l16) * BK], half);
      #pragma unroll
      for (int mt = 0; mt < 2; ++mt)
        acc[mt][nt] = __builtin_amdgcn_wmma_f32_16x16x32_bf16(
            false, af[mt], false, bv, (short)0, acc[mt][nt], false, false);
    }
    __syncthreads();
  }

  #pragma unroll
  for (int mt = 0; mt < 2; ++mt)
    #pragma unroll
    for (int nt = 0; nt < 4; ++nt) {
      int col = n0 + wn * 64 + nt * 16 + l16;
      #pragma unroll
      for (int i = 0; i < 8; ++i) {
        int row = m0 + wm * 32 + mt * 16 + i + 8 * half;
        float v = acc[mt][nt][i];
        if constexpr (sizeof(TC) == 4)
          C[(size_t)row * N + col] = v;
        else
          C[(size_t)row * N + col] = (bf16_t)v;
      }
    }
}

// ---------------------------------------------------------------------------
// RoPE (in-place, bf16 buffer [b*s][H*64]); oscale folds the attention
// softmax scale (0.125 * log2(e)) into Q.  total = B*S*H*32 pairs.
// ---------------------------------------------------------------------------
__global__ void rope_kernel(bf16_t* __restrict__ buf,
                            const float* __restrict__ cosb,
                            const float* __restrict__ sinb, int H, int total,
                            float oscale) {
  int idx = blockIdx.x * blockDim.x + threadIdx.x;
  if (idx >= total) return;
  int i = idx & 31;
  int h = (idx >> 5) % H;
  int sl = idx / (32 * H);  // b*S + s
  int s = sl & (S_ - 1);
  float c = cosb[s * 32 + i], sn = sinb[s * 32 + i];
  size_t base = (size_t)sl * (H * 64) + h * 64 + 2 * i;
  float xr = (float)buf[base], xi = (float)buf[base + 1];
  buf[base]     = (bf16_t)((xr * c - xi * sn) * oscale);
  buf[base + 1] = (bf16_t)((xr * sn + xi * c) * oscale);
}

// ---------------------------------------------------------------------------
// Causal GQA flash-attention, 64-key tiles.
// grid = (S/64, HQ, B), block = 128 threads = 4 waves; wave owns 16 q rows.
// Q pre-scaled by 0.125*log2(e) -> probabilities via exp2.
// Only the diagonal 64-key block needs masks; fully-masked score tiles are
// skipped with wave-uniform (readfirstlane) branches so EXEC stays all-ones
// for WMMA.  K tile goes to LDS via async-to-LDS when available.
// ---------------------------------------------------------------------------
__global__ __launch_bounds__(128) void attn_kernel(const bf16_t* __restrict__ Q,
                                                   const bf16_t* __restrict__ Kb,
                                                   const bf16_t* __restrict__ Vb,
                                                   bf16_t* __restrict__ O) {
  const int qblk = blockIdx.x;
  const int h    = blockIdx.y;
  const int b    = blockIdx.z;
  const int kvh  = h >> 2;
  const int tid  = threadIdx.x;
  const int lane = tid & 31;
  const int wid  = __builtin_amdgcn_readfirstlane(tid >> 5);  // uniform wave id
  const int l16  = lane & 15;
  const int half = lane >> 4;

  __shared__ __align__(16) bf16_t Ksm[64 * 64];    // [key][dim]
  __shared__ __align__(16) bf16_t Vsm[64 * 64];    // [dim][key]
  __shared__ __align__(16) bf16_t Psm[4][16 * 64]; // per-wave P staging

  const int qbase = qblk * 64 + wid * 16;
  const bf16_t* qp = &Q[((size_t)(b * S_) + qbase + l16) * (HQ_ * HD_) + h * HD_];
  v16bf aq[2];
  #pragma unroll
  for (int s = 0; s < 2; ++s) aq[s] = load_frag(qp + s * 32, half);

  v8f o[4] = {};
  float mrow[8], lrow[8];
  #pragma unroll
  for (int i = 0; i < 8; ++i) { mrow[i] = -__builtin_inff(); lrow[i] = 0.f; }

  for (int t = 0; t <= qblk; ++t) {
    // ---- cooperative K/V tile load: 64 keys x 64 dims each ----
    #pragma unroll
    for (int i = 0; i < 4; ++i) {
      int e = i * 1024 + tid * 8;
      int key = e >> 6, dim = e & 63;
      size_t src =
          ((size_t)(b * S_) + t * 64 + key) * (HKV_ * HD_) + kvh * HD_ + dim;
#if defined(USE_ASYNC_LDS)
      async_ld_b128(&Kb[src], &Ksm[key * 64 + dim]);  // no VGPR round-trip
#else
      *(bf16x8*)&Ksm[key * 64 + dim] = *(const bf16x8*)&Kb[src];
#endif
      bf16x8 vv = *(const bf16x8*)&Vb[src];           // V needs a transpose
      #pragma unroll
      for (int j = 0; j < 8; ++j) Vsm[(dim + j) * 64 + key] = vv[j];
    }
#if defined(USE_ASYNC_LDS)
    __builtin_amdgcn_s_wait_asynccnt(0);
#endif
    __syncthreads();

    const bool diag  = (t == qblk);
    const int  ktmax = diag ? wid : 3;   // skip fully-masked score tiles

    // ---- scores: up to 4 C tiles of 16q x 16k, contraction HD=64 ----
    v8f sc[4];
    #pragma unroll
    for (int kt = 0; kt < 4; ++kt) {
      if (kt > ktmax) continue;  // wave-uniform
      v8f c = {};
      #pragma unroll
      for (int s = 0; s < 2; ++s) {
        v16bf bk = load_frag(&Ksm[(kt * 16 + l16) * 64 + s * 32], half);
        c = __builtin_amdgcn_wmma_f32_16x16x32_bf16(false, aq[s], false, bk,
                                                    (short)0, c, false, false);
      }
      if (diag && kt == ktmax) {  // triangular mask, t-independent condition
        #pragma unroll
        for (int i = 0; i < 8; ++i)
          if (l16 > i + 8 * half) c[i] = -__builtin_inff();
      }
      sc[kt] = c;
    }

    // ---- online softmax (log2 domain; Q carries 0.125*log2e) ----
    float mnew[8], alpha[8];
    #pragma unroll
    for (int i = 0; i < 8; ++i) {
      float r = sc[0][i];
      #pragma unroll
      for (int kt = 1; kt < 4; ++kt)
        if (kt <= ktmax) r = fmaxf(r, sc[kt][i]);
      r        = row_max16(r);              // DPP, no DS traffic
      mnew[i]  = fmaxf(mrow[i], r);
      alpha[i] = exp2f(mrow[i] - mnew[i]);  // first block: exp2(-inf)=0
      mrow[i]  = mnew[i];
    }
    #pragma unroll
    for (int i = 0; i < 8; ++i) {
      float s = 0.f;
      #pragma unroll
      for (int kt = 0; kt < 4; ++kt)
        if (kt <= ktmax) {
          float p = exp2f(sc[kt][i] - mnew[i]);
          sc[kt][i] = p;
          s += p;
        }
      s = row_sum16(s);                     // DPP, no DS traffic
      lrow[i] = lrow[i] * alpha[i] + s;
    }
    #pragma unroll
    for (int n = 0; n < 4; ++n)
      #pragma unroll
      for (int i = 0; i < 8; ++i) o[n][i] *= alpha[i];

    // ---- stage P (C-layout -> A-layout), zeroing skipped tiles ----
    bf16_t* pw = &Psm[wid][0];
    #pragma unroll
    for (int i = 0; i < 8; ++i) {
      int row = i + 8 * half;
      #pragma unroll
      for (int kt = 0; kt < 4; ++kt)
        pw[row * 64 + kt * 16 + l16] =
            (bf16_t)(kt <= ktmax ? sc[kt][i] : 0.f);
    }

    // ---- O += P V ; drop the zero 32-key half on diagonal low waves ----
    const int ns = diag ? (wid >> 1) + 1 : 2;  // # of 32-key halves
    v16bf apf[2];
    #pragma unroll
    for (int s = 0; s < 2; ++s)
      if (s < ns) apf[s] = load_frag(&Psm[wid][l16 * 64 + s * 32], half);
    #pragma unroll
    for (int n = 0; n < 4; ++n)
      #pragma unroll
      for (int s = 0; s < 2; ++s)
        if (s < ns) {
          v16bf bv = load_frag(&Vsm[(n * 16 + l16) * 64 + s * 32], half);
          o[n] = __builtin_amdgcn_wmma_f32_16x16x32_bf16(
              false, apf[s], false, bv, (short)0, o[n], false, false);
        }
    __syncthreads();
  }

  // ---- epilogue: normalize, store [b][s][h][d] bf16 ----
  #pragma unroll
  for (int i = 0; i < 8; ++i) {
    float inv = 1.0f / lrow[i];
    int row = qbase + i + 8 * half;
    #pragma unroll
    for (int n = 0; n < 4; ++n)
      O[((size_t)(b * S_) + row) * (HQ_ * HD_) + h * HD_ + n * 16 + l16] =
          (bf16_t)(o[n][i] * inv);
  }
}

// ---------------------------------------------------------------------------
extern "C" void kernel_launch(void* const* d_in, const int* in_sizes, int n_in,
                              void* d_out, int out_size, void* d_ws,
                              size_t ws_size, hipStream_t stream) {
  const float* x  = (const float*)d_in[0];
  const float* fc = (const float*)d_in[1];
  const float* fs = (const float*)d_in[2];
  const float* wq = (const float*)d_in[3];
  const float* wk = (const float*)d_in[4];
  const float* wv = (const float*)d_in[5];
  const float* wo = (const float*)d_in[6];
  float* out = (float*)d_out;

  const int M = B_ * S_;  // 4096
  char* ws = (char*)d_ws;
  size_t off = 0;
  auto carve = [&](size_t bytes) {
    char* p = ws + off;
    off = (off + bytes + 255) & ~(size_t)255;
    return p;
  };
  bf16_t* qbuf = (bf16_t*)carve((size_t)M * HQ_ * HD_ * 2);   // 16.8 MB
  bf16_t* kbuf = (bf16_t*)carve((size_t)M * HKV_ * HD_ * 2);  //  4.2 MB
  bf16_t* vbuf = (bf16_t*)carve((size_t)M * HKV_ * HD_ * 2);  //  4.2 MB
  bf16_t* abuf = (bf16_t*)carve((size_t)M * HQ_ * HD_ * 2);   // 16.8 MB
  (void)ws_size;

  // 1) projections (bf16 out)
  gemm_wmma<float, bf16_t><<<dim3((HQ_ * HD_) / 128, M / 128), 256, 0, stream>>>(
      x, wq, qbuf, M, HQ_ * HD_, D_);
  gemm_wmma<float, bf16_t><<<dim3((HKV_ * HD_) / 128, M / 128), 256, 0, stream>>>(
      x, wk, kbuf, M, HKV_ * HD_, D_);
  gemm_wmma<float, bf16_t><<<dim3((HKV_ * HD_) / 128, M / 128), 256, 0, stream>>>(
      x, wv, vbuf, M, HKV_ * HD_, D_);

  // 2) RoPE; fold softmax scale * log2(e) into Q
  const float qscale = 0.125f * 1.44269504088896340736f;
  {
    int tq = M * HQ_ * 32;
    rope_kernel<<<(tq + 255) / 256, 256, 0, stream>>>(qbuf, fc, fs, HQ_, tq,
                                                      qscale);
    int tk = M * HKV_ * 32;
    rope_kernel<<<(tk + 255) / 256, 256, 0, stream>>>(kbuf, fc, fs, HKV_, tk,
                                                      1.0f);
  }

  // 3) attention
  attn_kernel<<<dim3(S_ / 64, HQ_, B_), 128, 0, stream>>>(qbuf, kbuf, vbuf,
                                                          abuf);

  // 4) output projection -> f32
  gemm_wmma<bf16_t, float><<<dim3(D_ / 128, M / 128), 256, 0, stream>>>(
      abuf, wo, out, M, D_, HQ_ * HD_);
}
